// JTVAE_48060684043091
// MI455X (gfx1250) — matrix operand order repaired and checked by reading, not compiled, not run on hardware
//
#include <hip/hip_runtime.h>
#include <math.h>

// ---------------------------------------------------------------- dims
#define HD   128      // hidden
#define FD   128      // node feat
#define ZD   56
#define VD   800
#define BD   64
#define MTN  12
#define NT_  3200
#define ET_  6400
#define NG_  50000
#define EG_  800000

typedef __attribute__((ext_vector_type(16))) __bf16 v16bf;
typedef __attribute__((ext_vector_type(8)))  float  v8f;

// native hardware convert (v_cvt on gfx1250) instead of integer RNE emulation
__device__ __forceinline__ __bf16 f2bf(float f) { return (__bf16)f; }

// ---------------------------------------------------------------- fill / convert
__global__ void k_fill0(float* p, int n) {
  int i = blockIdx.x * blockDim.x + threadIdx.x;
  if (i < n) p[i] = 0.f;
}
__global__ void k_cvt_bf16(const float* __restrict__ in, __bf16* __restrict__ out, int n) {
  int i = blockIdx.x * blockDim.x + threadIdx.x;
  if (i < n) out[i] = f2bf(in[i]);
}

// ---------------------------------------------------------------- A-fragment load
// 16-bit A layout: lane -> M row (lane&15); lanes<16 hold K 0-7/16-23,
// lanes>=16 hold K 8-15/24-31 of each 32-wide k-step.
__device__ __forceinline__ void load_afrag(const float* __restrict__ arow,
                                           int koff, v16bf afrag[4])
{
  #pragma unroll
  for (int ks = 0; ks < 4; ++ks) {
    const int kb = ks * 32 + koff;
    const float4 x0 = *(const float4*)(arow + kb);
    const float4 x1 = *(const float4*)(arow + kb + 4);
    const float4 y0 = *(const float4*)(arow + kb + 16);
    const float4 y1 = *(const float4*)(arow + kb + 20);
    afrag[ks][0]  = f2bf(x0.x); afrag[ks][1]  = f2bf(x0.y);
    afrag[ks][2]  = f2bf(x0.z); afrag[ks][3]  = f2bf(x0.w);
    afrag[ks][4]  = f2bf(x1.x); afrag[ks][5]  = f2bf(x1.y);
    afrag[ks][6]  = f2bf(x1.z); afrag[ks][7]  = f2bf(x1.w);
    afrag[ks][8]  = f2bf(y0.x); afrag[ks][9]  = f2bf(y0.y);
    afrag[ks][10] = f2bf(y0.z); afrag[ks][11] = f2bf(y0.w);
    afrag[ks][12] = f2bf(y1.x); afrag[ks][13] = f2bf(y1.y);
    afrag[ks][14] = f2bf(y1.z); afrag[ks][15] = f2bf(y1.w);
  }
}

// ---------------------------------------------------------------- WMMA GEMM
// C[M,128] = act(A[M,128] @ W[128,128] + bias).  A fp32 (cvt in-reg), W bf16.
// All M values are multiples of 16 -> no per-lane bounds checks; uniform
// wave-level early-out covers the grid remainder.
template <int RELU, int HAS_BIAS>
__global__ __launch_bounds__(128) void k_gemm_bf16(
    const float* __restrict__ A, const __bf16* __restrict__ W,
    const float* __restrict__ bias, float* __restrict__ C, int M)
{
  const int lane = threadIdx.x & 31;
  const int wave = threadIdx.x >> 5;
  const int m0 = (blockIdx.x * 4 + wave) * 16;
  if (m0 >= M) return;                       // uniform per wave

  const int half = lane >> 4;
  const float* arow = A + (size_t)(m0 + (lane & 15)) * HD;
  v16bf afrag[4];
  load_afrag(arow, half * 8, afrag);

  #pragma unroll
  for (int nt = 0; nt < HD / 16; ++nt) {
    v8f acc = {};
    #pragma unroll
    for (int ks = 0; ks < 4; ++ks) {
      const __bf16* wrow = W + (size_t)(ks * 32 + half * 8) * HD + nt * 16 + (lane & 15); // lane = N col
      v16bf bfrag;
      #pragma unroll
      for (int j = 0; j < 16; ++j) bfrag[j] = wrow[(size_t)(j < 8 ? j : j + 8) * HD];
      acc = __builtin_amdgcn_wmma_f32_16x16x32_bf16(
          false, afrag[ks], false, bfrag, (short)0, acc, false, false);
    }
    const int col = nt * 16 + (lane & 15);
    const float bv = HAS_BIAS ? bias[col] : 0.f;
    float* crow = C + (size_t)(m0 + half * 8) * HD + col;
    #pragma unroll
    for (int r = 0; r < 8; ++r) {            // C/D: VGPR r, +8 rows for upper lanes
      float v = acc[r] + bv;
      if (RELU) v = fmaxf(v, 0.f);
      crow[(size_t)r * HD] = v;
    }
  }
}

// Dual GEMM: Pd = A @ W0, Ps = A @ W1 (no bias/relu). A fragments loaded and
// converted ONCE, streamed through both weight matrices (64 WMMAs / wave).
__global__ __launch_bounds__(128) void k_gemm_dual_bf16(
    const float* __restrict__ A, const __bf16* __restrict__ W0,
    const __bf16* __restrict__ W1, float* __restrict__ C0,
    float* __restrict__ C1, int M)
{
  const int lane = threadIdx.x & 31;
  const int wave = threadIdx.x >> 5;
  const int m0 = (blockIdx.x * 4 + wave) * 16;
  if (m0 >= M) return;

  const int half = lane >> 4;
  const float* arow = A + (size_t)(m0 + (lane & 15)) * HD;
  v16bf afrag[4];
  load_afrag(arow, half * 8, afrag);

  const __bf16* Ws[2] = { W0, W1 };
  float*        Cs[2] = { C0, C1 };
  #pragma unroll
  for (int p = 0; p < 2; ++p) {
    const __bf16* W = Ws[p];
    float* C = Cs[p];
    #pragma unroll
    for (int nt = 0; nt < HD / 16; ++nt) {
      v8f acc = {};
      #pragma unroll
      for (int ks = 0; ks < 4; ++ks) {
        const __bf16* wrow = W + (size_t)(ks * 32 + half * 8) * HD + nt * 16 + (lane & 15);
        v16bf bfrag;
        #pragma unroll
        for (int j = 0; j < 16; ++j) bfrag[j] = wrow[(size_t)(j < 8 ? j : j + 8) * HD];
        acc = __builtin_amdgcn_wmma_f32_16x16x32_bf16(
            false, afrag[ks], false, bfrag, (short)0, acc, false, false);
      }
      const int col = nt * 16 + (lane & 15);
      float* crow = C + (size_t)(m0 + half * 8) * HD + col;
      #pragma unroll
      for (int r = 0; r < 8; ++r) crow[(size_t)r * HD] = acc[r];
    }
  }
}

// ---------------------------------------------------------------- edge message + scatter
// aggr[dst] += relu(Pd[dst] + Ps[src] + bias), 32 lanes x float4 per edge.
__global__ void k_edge(const float* __restrict__ Pd, const float* __restrict__ Ps,
                       const float* __restrict__ bias, const int* __restrict__ src,
                       const int* __restrict__ dst, float* __restrict__ aggr, int E)
{
  int idx = blockIdx.x * blockDim.x + threadIdx.x;
  int e = idx >> 5;
  if (e >= E) return;
  int c = (idx & 31) * 4;
  int s = src[e], d = dst[e];
  float4 pd = *(const float4*)(Pd + (size_t)d * HD + c);
  float4 ps = *(const float4*)(Ps + (size_t)s * HD + c);
  float4 bb = *(const float4*)(bias + c);
  float* ap = aggr + (size_t)d * HD + c;
  atomicAdd(ap + 0, fmaxf(pd.x + ps.x + bb.x, 0.f));
  atomicAdd(ap + 1, fmaxf(pd.y + ps.y + bb.y, 0.f));
  atomicAdd(ap + 2, fmaxf(pd.z + ps.z + bb.z, 0.f));
  atomicAdd(ap + 3, fmaxf(pd.w + ps.w + bb.w, 0.f));
}

// ---------------------------------------------------------------- mean pool
__global__ void k_pool(const float* __restrict__ h, const int* __restrict__ batch,
                       float* __restrict__ sums, float* __restrict__ cnt, int N)
{
  int idx = blockIdx.x * blockDim.x + threadIdx.x;
  int n = idx >> 5;
  if (n >= N) return;
  int c = (idx & 31) * 4;
  int b = batch[n];
  float4 v = *(const float4*)(h + (size_t)n * HD + c);
  float* sp = sums + b * HD + c;
  atomicAdd(sp + 0, v.x); atomicAdd(sp + 1, v.y);
  atomicAdd(sp + 2, v.z); atomicAdd(sp + 3, v.w);
  if (c == 0) atomicAdd(&cnt[b], 1.0f);
}
__global__ void k_pool_fin(const float* sums, const float* cnt, float* vec, int colOff) {
  int idx = blockIdx.x * blockDim.x + threadIdx.x;   // 64*128
  if (idx >= BD * HD) return;
  int b = idx >> 7, c = idx & 127;
  vec[b * (2 * HD) + colOff + c] = sums[b * HD + c] / fmaxf(cnt[b], 1.0f);
}

// ---------------------------------------------------------------- small dense head
__global__ void k_dense(const float* __restrict__ A, const float* __restrict__ W,
                        const float* __restrict__ bias, float* __restrict__ out,
                        int M, int N, int K, int relu)
{
  int idx = blockIdx.x * blockDim.x + threadIdx.x;
  if (idx >= M * N) return;
  int m = idx / N, n = idx % N;
  float s = bias ? bias[n] : 0.f;
  const float* a = A + (size_t)m * K;
  for (int k = 0; k < K; ++k) s += a[k] * W[(size_t)k * N + n];
  if (relu) s = fmaxf(s, 0.f);
  out[idx] = s;
}

__global__ void k_reparam(const float* mu, const float* lv, const float* eps, float* z, int n) {
  int i = blockIdx.x * blockDim.x + threadIdx.x;
  if (i < n) z[i] = mu[i] + eps[i] * expf(0.5f * lv[i]);
}

// logits = t @ fp2_w + fp2_b ; broadcast to 12 tree slots
__global__ void k_logits_bcast(const float* __restrict__ t, const float* __restrict__ W,
                               const float* __restrict__ bias, float* __restrict__ out)
{
  int idx = blockIdx.x * blockDim.x + threadIdx.x;   // 64*800
  if (idx >= BD * VD) return;
  int b = idx / VD, v = idx % VD;
  float s = bias[v];
  const float* a = t + b * HD;
  for (int k = 0; k < HD; ++k) s += a[k] * W[(size_t)k * VD + v];
  float* o = out + (size_t)b * MTN * VD + v;
  #pragma unroll
  for (int tt = 0; tt < MTN; ++tt) o[tt * VD] = s;
}

// ================================================================ host side
struct EncP {
  const float *proj_w, *proj_b;
  const float *msg_w[3], *msg_b[3], *lin_w[3], *lin_b[3];
};

static inline int ceil_div(int a, int b) { return (a + b - 1) / b; }

extern "C" void kernel_launch(void* const* d_in, const int* in_sizes, int n_in,
                              void* d_out, int out_size, void* d_ws, size_t ws_size,
                              hipStream_t stream)
{
  const float* tree_x  = (const float*)d_in[0];
  const int*   tree_ei = (const int*)d_in[1];
  const float* graph_x = (const float*)d_in[2];
  const int*   graph_ei= (const int*)d_in[3];
  const int*   batch_t = (const int*)d_in[4];
  const int*   batch_g = (const int*)d_in[5];
  const float* eps     = (const float*)d_in[6];

  // ---- param pointers (JAX pytree: sorted dict keys, list order)
  const float *fp1_b,*fp1_w,*fp2_b,*fp2_w,*nd_b,*nd_w,*z2h_b,*z2h_w;
  const float *fc_lv_b,*fc_lv_w,*fc_mu_b,*fc_mu_w;
  EncP ge, te;
  if (n_in >= 48) {                       // separate leaves at d_in[8..47]
    const float* const* L = (const float* const*)(d_in + 8);
    fp1_b=L[0]; fp1_w=L[1]; fp2_b=L[2]; fp2_w=L[3];
    nd_b =L[4]; nd_w =L[5]; z2h_b=L[6]; z2h_w=L[7];
    fc_lv_b=L[8]; fc_lv_w=L[9]; fc_mu_b=L[10]; fc_mu_w=L[11];
    for (int l = 0; l < 3; ++l) {
      ge.lin_b[l]=L[12+4*l]; ge.lin_w[l]=L[13+4*l]; ge.msg_b[l]=L[14+4*l]; ge.msg_w[l]=L[15+4*l];
      te.lin_b[l]=L[26+4*l]; te.lin_w[l]=L[27+4*l]; te.msg_b[l]=L[28+4*l]; te.msg_w[l]=L[29+4*l];
    }
    ge.proj_b=L[24]; ge.proj_w=L[25]; te.proj_b=L[38]; te.proj_w=L[39];
  } else {                                // one concatenated blob at d_in[8]
    const float* P = (const float*)d_in[8];
    fp1_b=P+0;       fp1_w=P+128;     fp2_b=P+16512;  fp2_w=P+17312;
    nd_b =P+119712;  nd_w =P+119744;  z2h_b=P+123840; z2h_w=P+123968;
    fc_lv_b=P+131136; fc_lv_w=P+131192; fc_mu_b=P+145528; fc_mu_w=P+145584;
    auto fill = [&](EncP& e, const float* base) {
      for (int l = 0; l < 3; ++l) {
        e.lin_b[l]=base+l*49408;       e.lin_w[l]=base+l*49408+128;
        e.msg_b[l]=base+l*49408+16512; e.msg_w[l]=base+l*49408+16640;
      }
      e.proj_b = base + 148224; e.proj_w = base + 148352;
    };
    fill(ge, P + 159920); fill(te, P + 324656);
  }

  // ---- workspace carve-up
  char* ws = (char*)d_ws;
  size_t off = 0;
  auto alloc = [&](size_t bytes) { char* p = ws + off; off = (off + bytes + 255) & ~(size_t)255; return p; };
  const size_t ENC_BF = 16384 + 3 * (32768 + 16384);          // bf16 elems / encoder
  __bf16* wbf_g = (__bf16*)alloc(ENC_BF * 2);
  __bf16* wbf_t = (__bf16*)alloc(ENC_BF * 2);
  float*  hbuf  = (float*)alloc((size_t)NG_ * HD * 4);
  float*  Pd    = (float*)alloc((size_t)NG_ * HD * 4);
  float*  Ps    = (float*)alloc((size_t)NG_ * HD * 4);
  float*  aggr  = (float*)alloc((size_t)NG_ * HD * 4);
  float*  sums  = (float*)alloc((size_t)BD * HD * 4);
  float*  cnt   = (float*)alloc((size_t)BD * 4);
  float*  vec   = (float*)alloc((size_t)BD * 2 * HD * 4);
  float*  zbuf  = (float*)alloc((size_t)BD * ZD * 4);
  float*  hdec  = (float*)alloc((size_t)BD * HD * 4);
  float*  tbuf  = (float*)alloc((size_t)BD * HD * 4);
  (void)ws_size; (void)in_sizes; (void)out_size;

  float* out = (float*)d_out;
  const int OFF_ND = BD * MTN * VD;           // 614400
  const int OFF_MU = OFF_ND + BD * 32;        // 616448
  const int OFF_LV = OFF_MU + BD * ZD;        // 620032

  auto cvt = [&](const float* src, __bf16* dst, int n) {
    k_cvt_bf16<<<ceil_div(n, 256), 256, 0, stream>>>(src, dst, n);
  };
  auto fill0 = [&](float* p, int n) {
    k_fill0<<<ceil_div(n, 256), 256, 0, stream>>>(p, n);
  };

  auto runEncoder = [&](const float* x, int N, const int* ei, int E,
                        const int* batch, const EncP& ep, __bf16* wbf, int colOff) {
    const int* src = ei;
    const int* dst = ei + E;
    // weights -> bf16 (proj | [msg,lin] x3)
    cvt(ep.proj_w, wbf, HD * HD);
    for (int l = 0; l < 3; ++l) {
      cvt(ep.msg_w[l], wbf + 16384 + l * 49152, 2 * HD * HD);
      cvt(ep.lin_w[l], wbf + 16384 + l * 49152 + 32768, HD * HD);
    }
    const int gblk = ceil_div(N, 64);
    k_gemm_bf16<1, 1><<<gblk, 128, 0, stream>>>(x, wbf, ep.proj_b, hbuf, N);
    for (int l = 0; l < 3; ++l) {
      const __bf16* msgbf = wbf + 16384 + l * 49152;
      const __bf16* linbf = msgbf + 32768;
      k_gemm_dual_bf16<<<gblk, 128, 0, stream>>>(hbuf, msgbf, msgbf + HD * HD, Pd, Ps, N);
      fill0(aggr, N * HD);
      k_edge<<<ceil_div(E * 32, 256), 256, 0, stream>>>(Pd, Ps, ep.msg_b[l], src, dst, aggr, E);
      k_gemm_bf16<1, 1><<<gblk, 128, 0, stream>>>(aggr, linbf, ep.lin_b[l], hbuf, N);
    }
    fill0(sums, BD * HD);
    fill0(cnt, BD);
    k_pool<<<ceil_div(N * 32, 256), 256, 0, stream>>>(hbuf, batch, sums, cnt, N);
    k_pool_fin<<<ceil_div(BD * HD, 256), 256, 0, stream>>>(sums, cnt, vec, colOff);
  };

  runEncoder(tree_x,  NT_, tree_ei,  ET_, batch_t, te, wbf_t, 0);
  runEncoder(graph_x, NG_, graph_ei, EG_, batch_g, ge, wbf_g, HD);

  // ---- VAE head (tiny)
  k_dense<<<ceil_div(BD * ZD, 256), 256, 0, stream>>>(vec, fc_mu_w, fc_mu_b, out + OFF_MU, BD, ZD, 2 * HD, 0);
  k_dense<<<ceil_div(BD * ZD, 256), 256, 0, stream>>>(vec, fc_lv_w, fc_lv_b, out + OFF_LV, BD, ZD, 2 * HD, 0);
  k_reparam<<<ceil_div(BD * ZD, 256), 256, 0, stream>>>(out + OFF_MU, out + OFF_LV, eps, zbuf, BD * ZD);
  k_dense<<<ceil_div(BD * HD, 256), 256, 0, stream>>>(zbuf, z2h_w, z2h_b, hdec, BD, HD, ZD, 1);
  k_dense<<<ceil_div(BD * HD, 256), 256, 0, stream>>>(hdec, fp1_w, fp1_b, tbuf, BD, HD, HD, 1);
  k_logits_bcast<<<ceil_div(BD * VD, 256), 256, 0, stream>>>(tbuf, fp2_w, fp2_b, out);
  k_dense<<<ceil_div(BD * 32, 256), 256, 0, stream>>>(hdec, nd_w, nd_b, out + OFF_ND, BD, 32, HD, 0);
}
